// Attention_2551210573833
// MI455X (gfx1250) — matrix operand order repaired and checked by reading, not compiled, not run on hardware
//
#include <hip/hip_runtime.h>
#include <hip/hip_bf16.h>

typedef __attribute__((ext_vector_type(16))) _Float16 v16h;
typedef __attribute__((ext_vector_type(8)))  _Float16 v8h;
typedef __attribute__((ext_vector_type(8)))  float    v8f;

#define B_   2
#define N_   1024
#define DIM_ 1024
#define H_   16
#define DH_  64
#define NKEY 1025          // null + N
#define KPAD 1056          // 33 * 32 key blocks
#define LNEPS 1e-5f

// ---------------------------------------------------------------------------
// WMMA fragment index helpers (CDNA5 ISA 7.12.2, wave32)
// A 16x32 f16:  lane lo = row m (0..15); element e -> K = e + 16*hi  (contiguous)
// B 32x16 f16:  lane lo = col n;         element e -> K = e + 16*hi  (contiguous)
// C 16x16 f32:  lane lo = col n; vgpr r -> row m = r + 8*hi
// ---------------------------------------------------------------------------

__device__ __forceinline__ v16h load_a_frag(const _Float16* row, int hi) {
    // two contiguous 8-half chunks: [hi*16 .. hi*16+7] and [hi*16+8 .. hi*16+15]
    v8h lo8 = *(const v8h*)(row + hi * 16);
    v8h hi8 = *(const v8h*)(row + hi * 16 + 8);
    v16h a;
#pragma unroll
    for (int e = 0; e < 8; ++e) { a[e] = lo8[e]; a[e + 8] = hi8[e]; }
    return a;
}

// ---------------------------------------------------------------------------
// LayerNorm input: x[row][1024] -> f16
// ---------------------------------------------------------------------------
__global__ void ln_in_kernel(const float* __restrict__ x, const float* __restrict__ g,
                             _Float16* __restrict__ xn) {
    const int row = blockIdx.x;
    const int tid = threadIdx.x;
    const float* xr = x + (size_t)row * DIM_;
    __shared__ float red[256];
    float s = 0.f;
    for (int c = tid; c < DIM_; c += 256) s += xr[c];
    red[tid] = s; __syncthreads();
    for (int w = 128; w > 0; w >>= 1) { if (tid < w) red[tid] += red[tid + w]; __syncthreads(); }
    const float mean = red[0] * (1.f / DIM_);
    __syncthreads();
    float vs = 0.f;
    for (int c = tid; c < DIM_; c += 256) { float d = xr[c] - mean; vs += d * d; }
    red[tid] = vs; __syncthreads();
    for (int w = 128; w > 0; w >>= 1) { if (tid < w) red[tid] += red[tid + w]; __syncthreads(); }
    const float rstd = rsqrtf(red[0] * (1.f / DIM_) + LNEPS);
    _Float16* orow = xn + (size_t)row * DIM_;
    for (int c = tid; c < DIM_; c += 256)
        orow[c] = (_Float16)((xr[c] - mean) * rstd * g[c]);
}

// LayerNorm output: f32 -> f32 d_out
__global__ void ln_out_kernel(const float* __restrict__ x, const float* __restrict__ g,
                              float* __restrict__ out) {
    const int row = blockIdx.x;
    const int tid = threadIdx.x;
    const float* xr = x + (size_t)row * DIM_;
    __shared__ float red[256];
    float s = 0.f;
    for (int c = tid; c < DIM_; c += 256) s += xr[c];
    red[tid] = s; __syncthreads();
    for (int w = 128; w > 0; w >>= 1) { if (tid < w) red[tid] += red[tid + w]; __syncthreads(); }
    const float mean = red[0] * (1.f / DIM_);
    __syncthreads();
    float vs = 0.f;
    for (int c = tid; c < DIM_; c += 256) { float d = xr[c] - mean; vs += d * d; }
    red[tid] = vs; __syncthreads();
    for (int w = 128; w > 0; w >>= 1) { if (tid < w) red[tid] += red[tid + w]; __syncthreads(); }
    const float rstd = rsqrtf(red[0] * (1.f / DIM_) + LNEPS);
    float* orow = out + (size_t)row * DIM_;
    for (int c = tid; c < DIM_; c += 256)
        orow[c] = (xr[c] - mean) * rstd * g[c];
}

// ---------------------------------------------------------------------------
// f32 -> f16 conversion (weights)
// ---------------------------------------------------------------------------
__global__ void cvt_f16_kernel(const float* __restrict__ src, _Float16* __restrict__ dst, int n) {
    int i = blockIdx.x * 256 + threadIdx.x;
    if (i < n) dst[i] = (_Float16)src[i];
}

// ---------------------------------------------------------------------------
// Generic WMMA GEMM: C[M x Nc] = A[M x K] * W[Nc x K]^T  (A,W f16; C f32)
// grid = (M/32, Nc/64); one wave per block computes a 32x64 tile:
// 2 A-fragments + 4 B-fragments feed 8 WMMAs per 32-wide k-step.
// ---------------------------------------------------------------------------
__global__ void gemm_f16_kernel(const _Float16* __restrict__ A, const _Float16* __restrict__ W,
                                float* __restrict__ C, int M, int Nc, int K) {
    const int i0 = blockIdx.x * 32;
    const int n0 = blockIdx.y * 64;
    const int lane = threadIdx.x;
    const int hi = lane >> 4, lo = lane & 15;

    v8f acc[2][4];
#pragma unroll
    for (int p = 0; p < 2; ++p)
#pragma unroll
        for (int t = 0; t < 4; ++t) acc[p][t] = v8f{};

    const _Float16* arow0 = A + (size_t)(i0 + lo) * K;
    const _Float16* arow1 = A + (size_t)(i0 + 16 + lo) * K;
    for (int k0 = 0; k0 < K; k0 += 32) {
        v16h a0 = load_a_frag(arow0 + k0, hi);
        v16h a1 = load_a_frag(arow1 + k0, hi);
#pragma unroll
        for (int t = 0; t < 4; ++t) {
            const _Float16* wrow = W + (size_t)(n0 + t * 16 + lo) * K + k0 + hi * 16;
            v16h b = *(const v16h*)wrow;   // 32B-aligned contiguous B fragment
            acc[0][t] = __builtin_amdgcn_wmma_f32_16x16x32_f16(
                false, a0, false, b, (short)0, acc[0][t], false, false);
            acc[1][t] = __builtin_amdgcn_wmma_f32_16x16x32_f16(
                false, a1, false, b, (short)0, acc[1][t], false, false);
        }
    }
#pragma unroll
    for (int p = 0; p < 2; ++p)
#pragma unroll
        for (int t = 0; t < 4; ++t)
#pragma unroll
            for (int r = 0; r < 8; ++r)
                C[(size_t)(i0 + p * 16 + r + hi * 8) * Nc + n0 + t * 16 + lo] = acc[p][t][r];
}

// ---------------------------------------------------------------------------
// Q prep: q_f32 [(b*N+i)*1024 + h*64 + d] -> l2norm * 4 -> q_f16 [(b*H+h)*N+i][64]
// one wave per row; each lane owns dims {lane, lane+32}
// ---------------------------------------------------------------------------
__global__ void prep_q_kernel(const float* __restrict__ qf32, _Float16* __restrict__ qf16) {
    const int row = blockIdx.x;              // (b*H+h)*N + i
    const int i = row & (N_ - 1);
    const int bh = row >> 10;
    const int b = bh / H_, h = bh % H_;
    const int lane = threadIdx.x;
    const float* src = qf32 + ((size_t)(b * N_ + i) * DIM_) + h * DH_;
    float v0 = src[lane], v1 = src[lane + 32];
    float ss = v0 * v0 + v1 * v1;
#pragma unroll
    for (int off = 1; off < 32; off <<= 1) ss += __shfl_xor(ss, off, 32);
    const float inv = 4.f / fmaxf(sqrtf(ss), 1e-12f);   // sqrt(SCALE)=4; pre-scale cancels in l2norm
    _Float16* dst = qf16 + ((size_t)bh * N_ + i) * DH_;
    dst[lane]      = (_Float16)(v0 * inv);
    dst[lane + 32] = (_Float16)(v1 * inv);
}

// ---------------------------------------------------------------------------
// K/V prep: kv_f32 [(b*N+j-1)*128], null_kv[2][64]; pad to KPAD keys.
// K stored row-major [b][key][dim] (contiguous B-fragments over dims);
// V stored TRANSPOSED [b][dim][KPAD] so attn's B-fragments over keys are
// contiguous 128-bit loads.
// ---------------------------------------------------------------------------
__global__ void prep_kv_kernel(const float* __restrict__ kvf32, const float* __restrict__ null_kv,
                               _Float16* __restrict__ kf, _Float16* __restrict__ vt) {
    const int idx = blockIdx.x;              // b*KPAD + j
    const int b = idx / KPAD, j = idx % KPAD;
    const int lane = threadIdx.x;
    float k0, k1, v0, v1;
    if (j == 0) {
        k0 = null_kv[lane];       k1 = null_kv[lane + 32];
        v0 = null_kv[64 + lane];  v1 = null_kv[64 + lane + 32];
    } else if (j <= N_) {
        const float* src = kvf32 + (size_t)(b * N_ + (j - 1)) * (2 * DH_);
        k0 = src[lane];       k1 = src[lane + 32];
        v0 = src[64 + lane];  v1 = src[64 + lane + 32];
    } else {
        k0 = k1 = v0 = v1 = 0.f;
    }
    float ss = k0 * k0 + k1 * k1;
#pragma unroll
    for (int off = 1; off < 32; off <<= 1) ss += __shfl_xor(ss, off, 32);
    const float inv = 4.f / fmaxf(sqrtf(ss), 1e-12f);
    _Float16* kd = kf + (size_t)idx * DH_;
    kd[lane]      = (_Float16)(k0 * inv);
    kd[lane + 32] = (_Float16)(k1 * inv);
    // V transposed: vt[(b*DH + d) * KPAD + j]
    vt[((size_t)b * DH_ + lane) * KPAD + j]        = (_Float16)v0;
    vt[((size_t)b * DH_ + lane + 32) * KPAD + j]   = (_Float16)v1;
}

// ---------------------------------------------------------------------------
// Flash attention: one wave per (b,h, 16-row tile). Streams bias (the HBM-bound
// term), WMMA for QK^T and P*V, online softmax in f32.
// ---------------------------------------------------------------------------
__global__ void attn_kernel(const _Float16* __restrict__ qf, const _Float16* __restrict__ kf,
                            const _Float16* __restrict__ vt, const float* __restrict__ bias,
                            _Float16* __restrict__ att) {
    const int tile = blockIdx.x;             // b*H*(N/16) tiles
    const int nt = N_ / 16;
    const int i0 = (tile % nt) * 16;
    const int bh = tile / nt;
    const int b = bh / H_, h = bh % H_;
    const int lane = threadIdx.x;
    const int hi = lane >> 4, lo = lane & 15;

    __shared__ _Float16 P[16][32];

    // Q fragments for the whole tile (d-chunks 0..31, 32..63)
    const _Float16* qrow = qf + ((size_t)bh * N_ + i0 + lo) * DH_;
    v16h qa0 = load_a_frag(qrow, hi);
    v16h qa1 = load_a_frag(qrow + 32, hi);

    v8f o[4];
#pragma unroll
    for (int t = 0; t < 4; ++t) o[t] = v8f{};
    float mrow[8], lrow[8];
#pragma unroll
    for (int r = 0; r < 8; ++r) { mrow[r] = -3e38f; lrow[r] = 0.f; }

    const float* brow = bias + ((size_t)bh * N_ + i0) * NKEY;
    const _Float16* kbase = kf + (size_t)b * KPAD * DH_;
    const _Float16* vbase = vt + (size_t)b * DH_ * KPAD;

    for (int j0 = 0; j0 < KPAD; j0 += 32) {
        // prefetch next key-block's bias stream (global_prefetch_b8)
        if (j0 + 32 < NKEY)
            __builtin_prefetch(brow + (size_t)(hi * 8) * NKEY + j0 + 32 + lo, 0, 1);

        v8f s[2];
#pragma unroll
        for (int jt = 0; jt < 2; ++jt) {
            s[jt] = v8f{};
            const int key = j0 + jt * 16 + lo;
            const _Float16* krow = kbase + (size_t)key * DH_ + hi * 16;
            v16h kb0 = *(const v16h*)(krow);        // d = 0..31 fragment (contiguous)
            v16h kb1 = *(const v16h*)(krow + 32);   // d = 32..63 fragment
            s[jt] = __builtin_amdgcn_wmma_f32_16x16x32_f16(false, qa0, false, kb0, (short)0, s[jt], false, false);
            s[jt] = __builtin_amdgcn_wmma_f32_16x16x32_f16(false, qa1, false, kb1, (short)0, s[jt], false, false);
            // + attn_bias (streamed once from HBM), pad keys -> -inf
            const bool in = (key < NKEY);
#pragma unroll
            for (int r = 0; r < 8; ++r) {
                const int m = r + hi * 8;
                float bv = in ? brow[(size_t)m * NKEY + key] : 0.f;
                s[jt][r] = in ? (s[jt][r] + bv) : -3e38f;
            }
        }
        // online softmax over this 32-key block (rows live across 16 lanes/half)
#pragma unroll
        for (int r = 0; r < 8; ++r) {
            const int m = r + hi * 8;
            float tmax = fmaxf(s[0][r], s[1][r]);
#pragma unroll
            for (int off = 1; off < 16; off <<= 1) tmax = fmaxf(tmax, __shfl_xor(tmax, off, 32));
            const float mnew = fmaxf(mrow[r], tmax);
            const float alpha = expf(mrow[r] - mnew);
            const float p0 = expf(s[0][r] - mnew);
            const float p1 = expf(s[1][r] - mnew);
            float psum = p0 + p1;
#pragma unroll
            for (int off = 1; off < 16; off <<= 1) psum += __shfl_xor(psum, off, 32);
            lrow[r] = lrow[r] * alpha + psum;
            mrow[r] = mnew;
#pragma unroll
            for (int t = 0; t < 4; ++t) o[t][r] *= alpha;
            P[m][lo]      = (_Float16)p0;
            P[m][16 + lo] = (_Float16)p1;
        }
        // refragment P (C layout -> A layout) via LDS; single wave => program order
        v16h pa = load_a_frag(&P[lo][0], hi);
        // V fragments from transposed V: per lane dim = t*16+lo, keys contiguous
#pragma unroll
        for (int t = 0; t < 4; ++t) {
            const _Float16* vrow = vbase + (size_t)(t * 16 + lo) * KPAD + j0 + hi * 16;
            v16h vb = *(const v16h*)vrow;  // 32B-aligned: 16 consecutive keys
            o[t] = __builtin_amdgcn_wmma_f32_16x16x32_f16(false, pa, false, vb, (short)0, o[t], false, false);
        }
    }

    // epilogue: normalize by row sums, store to att[b*N+i][h*64+d] f16
#pragma unroll
    for (int r = 0; r < 8; ++r) {
        const int m = r + hi * 8;
        const float inv = 1.f / lrow[r];
        _Float16* orow = att + ((size_t)(b * N_ + i0 + m)) * (H_ * DH_) + h * DH_;
#pragma unroll
        for (int t = 0; t < 4; ++t)
            orow[t * 16 + lo] = (_Float16)(o[t][r] * inv);
    }
}

// ---------------------------------------------------------------------------
// Launch
// ---------------------------------------------------------------------------
extern "C" void kernel_launch(void* const* d_in, const int* in_sizes, int n_in,
                              void* d_out, int out_size, void* d_ws, size_t ws_size,
                              hipStream_t stream) {
    const float* x       = (const float*)d_in[0];
    const float* bias    = (const float*)d_in[1];
    const float* g_in    = (const float*)d_in[2];
    const float* w_q     = (const float*)d_in[3];
    const float* w_kv    = (const float*)d_in[4];
    const float* null_kv = (const float*)d_in[5];
    const float* w_out   = (const float*)d_in[6];
    const float* g_out   = (const float*)d_in[7];
    float* out = (float*)d_out;

    char* ws = (char*)d_ws;
    size_t off = 0;
    auto carve = [&](size_t bytes) -> void* {
        void* p = ws + off;
        off += (bytes + 255) & ~(size_t)255;
        return p;
    };
    const size_t ROWS = (size_t)B_ * N_;                       // 2048
    _Float16* xn16   = (_Float16*)carve(ROWS * DIM_ * 2);      // 4 MB
    _Float16* wq16   = (_Float16*)carve((size_t)DIM_ * DIM_ * 2);
    _Float16* wkv16  = (_Float16*)carve((size_t)2 * DH_ * DIM_ * 2);
    _Float16* wout16 = (_Float16*)carve((size_t)DIM_ * DIM_ * 2);
    float*    q32    = (float*)carve(ROWS * DIM_ * 4);         // 8 MB
    float*    kv32   = (float*)carve(ROWS * 2 * DH_ * 4);      // 1 MB
    _Float16* q16    = (_Float16*)carve((size_t)B_ * H_ * N_ * DH_ * 2);
    _Float16* k16    = (_Float16*)carve((size_t)B_ * KPAD * DH_ * 2);
    _Float16* vt16   = (_Float16*)carve((size_t)B_ * DH_ * KPAD * 2);
    _Float16* att16  = (_Float16*)carve(ROWS * DIM_ * 2);      // 4 MB
    float*    out32  = (float*)carve(ROWS * DIM_ * 4);         // 8 MB

    // 1) input layernorm -> f16
    ln_in_kernel<<<ROWS, 256, 0, stream>>>(x, g_in, xn16);
    // 2) weights -> f16
    cvt_f16_kernel<<<(DIM_ * DIM_) / 256, 256, 0, stream>>>(w_q, wq16, DIM_ * DIM_);
    cvt_f16_kernel<<<(2 * DH_ * DIM_) / 256, 256, 0, stream>>>(w_kv, wkv16, 2 * DH_ * DIM_);
    cvt_f16_kernel<<<(DIM_ * DIM_) / 256, 256, 0, stream>>>(w_out, wout16, DIM_ * DIM_);
    // 3) projections (WMMA)
    gemm_f16_kernel<<<dim3(ROWS / 32, DIM_ / 64), 32, 0, stream>>>(xn16, wq16, q32, ROWS, DIM_, DIM_);
    gemm_f16_kernel<<<dim3(ROWS / 32, (2 * DH_) / 64), 32, 0, stream>>>(xn16, wkv16, kv32, ROWS, 2 * DH_, DIM_);
    // 4) q/k/v prep (l2norm, null key, padding, V transpose)
    prep_q_kernel<<<B_ * H_ * N_, 32, 0, stream>>>(q32, q16);
    prep_kv_kernel<<<B_ * KPAD, 32, 0, stream>>>(kv32, null_kv, k16, vt16);
    // 5) flash attention (WMMA)
    attn_kernel<<<B_ * H_ * (N_ / 16), 32, 0, stream>>>(q16, k16, vt16, bias, att16);
    // 6) output projection (WMMA)
    gemm_f16_kernel<<<dim3(ROWS / 32, DIM_ / 64), 32, 0, stream>>>(att16, wout16, out32, ROWS, DIM_, DIM_);
    // 7) output layernorm -> f32
    ln_out_kernel<<<ROWS, 256, 0, stream>>>(out32, g_out, out);
}
